// LinearAttention_68410239091435
// MI455X (gfx1250) — matrix operand order repaired and checked by reading, not compiled, not run on hardware
//
#include <hip/hip_runtime.h>
#include <hip/hip_bf16.h>
#include <math.h>
#include <stdint.h>

typedef __attribute__((ext_vector_type(16))) __bf16 v16bf;
typedef __attribute__((ext_vector_type(8)))  __bf16 v8bf;
typedef __attribute__((ext_vector_type(8)))  float  v8f;
typedef __attribute__((ext_vector_type(4)))  int    v4i;

#define AS1 __attribute__((address_space(1)))
#define AS3 __attribute__((address_space(3)))

#define D_MODEL 1024
#define T_SEQ   1024
#define NBATCH  4
#define NH      16
#define DH      64

#define KC      128          // k-chunk staged in LDS
#define XSTRIDE (KC + 8)     // +16B pad to spread LDS banks across rows

#if __has_builtin(__builtin_amdgcn_global_load_async_to_lds_b128)
#define HAVE_ASYNC_LDS 1
#endif

__device__ __forceinline__ v16bf cat8(v8bf lo, v8bf hi) {
  return __builtin_shufflevector(lo, hi, 0,1,2,3,4,5,6,7,8,9,10,11,12,13,14,15);
}

// A-operand (16x32 bf16, lane owns row m=lane&15):
// elements K = k0+8*half+{0..7}  and  k0+16+8*half+{0..7}
__device__ __forceinline__ v16bf frag_a_bf16(const __bf16* row, int k0, int half) {
  v8bf lo = *(const v8bf*)(row + k0 + half * 8);
  v8bf hi = *(const v8bf*)(row + k0 + 16 + half * 8);
  return cat8(lo, hi);
}
// B-operand (32x16 bf16, lane owns col n=lane&15):
// elements K = k0+8*half+{0..7}  and  k0+16+8*half+{0..7}
__device__ __forceinline__ v16bf frag_b_bf16(const __bf16* row, int k0, int half) {
  const __bf16* p = row + k0 + half * 8;
  return cat8(*(const v8bf*)p, *(const v8bf*)(p + 16));
}

// ---- async global->LDS staging helpers ----------------------------------------
__device__ __forceinline__ void copy16_g2l(__bf16* l, const __bf16* g) {
#ifdef HAVE_ASYNC_LDS
  // signature (from clang diagnostic): (AS1 v4i* gsrc, AS3 v4i* ldst, Ii off, Ii cpol)
  __builtin_amdgcn_global_load_async_to_lds_b128(
      (AS1 v4i*)(uintptr_t)g,   // global AS shares the generic VA: int round-trip is exact
      (AS3 v4i*)l,              // addrspacecast generic -> LDS
      0, 0);
#else
  *(v8bf*)l = *(const v8bf*)g;  // fallback: vgpr bounce, lowers to ds_store_b128
#endif
}
__device__ __forceinline__ void wait_async0() {
#ifdef HAVE_ASYNC_LDS
  asm volatile("s_wait_asynccnt 0x0" ::: "memory");
#endif
}
__device__ __forceinline__ void wait_async1() {
#ifdef HAVE_ASYNC_LDS
  asm volatile("s_wait_asynccnt 0x1" ::: "memory");
#endif
}

// Stage one 16 x 128 bf16 chunk of the activation tile: 256 threads x one b128.
__device__ __forceinline__ void stage_chunk(__bf16 (*xt)[16][XSTRIDE],
                                            int buf, const __bf16* Ag,
                                            int m0, int c, int tid) {
  const int row = tid >> 4, cb = tid & 15;
  const __bf16* g = Ag + (size_t)(m0 + row) * D_MODEL + c * KC + cb * 8;
  copy16_g2l(&xt[buf][row][cb * 8], g);
}

// ---------------- Kernel 0: one-shot fp32 -> bf16 conversion -------------------
__global__ __launch_bounds__(256)
void cvt_bf16_kernel(const float* __restrict__ s, __bf16* __restrict__ d, int n4) {
  int i = blockIdx.x * blockDim.x + threadIdx.x;
  if (i < n4) {
    float4 v = ((const float4*)s)[i];
    __align__(8) __bf16 o[4] = {(__bf16)v.x, (__bf16)v.y, (__bf16)v.z, (__bf16)v.w};
    ((uint2*)d)[i] = *(const uint2*)o;
  }
}

// ---------------- Kernel 1: fused QKV projection + RoPE, V stored transposed ----
// Block: 8 waves share one 16-row x-tile (LDS-staged, async double-buffered);
// each wave owns a 16x64 output tile (one full head) = 4 WMMA accumulators.
__global__ __launch_bounds__(256)
void qkv_rope_kernel(const __bf16* __restrict__ Xb,
                     const __bf16* __restrict__ Wqb,
                     const __bf16* __restrict__ Wkb,
                     const __bf16* __restrict__ Wvb,
                     __bf16* __restrict__ Qb,   // [B,H,T,DH]
                     __bf16* __restrict__ Kb,   // [B,H,T,DH]
                     __bf16* __restrict__ Vt) { // [B,H,DH,T]
  __shared__ __align__(16) __bf16 xt[2][16][XSTRIDE];

  const int tid  = threadIdx.x;
  const int lane = tid & 31;
  const int wave = tid >> 5;
  const int half = lane >> 4;
  const int ln   = lane & 15;

  const int m0   = blockIdx.x * 16;                 // row tile base in [0,4096)
  const int og   = (blockIdx.y * 8 + wave) * 64;    // out-col tile base in [0,3072)
  const int proj = og >> 10;                        // 0=Q,1=K,2=V
  const int o_in = og & (D_MODEL - 1);

  const __bf16* W = (proj == 0) ? Wqb : (proj == 1 ? Wkb : Wvb);
  const __bf16* wrow0 = W + (size_t)(o_in +  0 + ln) * D_MODEL;  // B[k][n]=W[n][k]
  const __bf16* wrow1 = W + (size_t)(o_in + 16 + ln) * D_MODEL;
  const __bf16* wrow2 = W + (size_t)(o_in + 32 + ln) * D_MODEL;
  const __bf16* wrow3 = W + (size_t)(o_in + 48 + ln) * D_MODEL;

  stage_chunk(xt, 0, Xb, m0, 0, tid);
  wait_async0();
  __syncthreads();
  stage_chunk(xt, 1, Xb, m0, 1, tid);

  v8f acc0 = {}, acc1 = {}, acc2 = {}, acc3 = {};
  for (int c = 0; c < 8; c++) {
    const __bf16* xrow = &xt[c & 1][ln][0];
#pragma unroll
    for (int kk = 0; kk < KC; kk += 32) {
      const int k = c * KC + kk;
      v16bf a = frag_a_bf16(xrow, kk, half);
      acc0 = __builtin_amdgcn_wmma_f32_16x16x32_bf16(false, a, false, frag_b_bf16(wrow0, k, half), (short)0, acc0, false, false);
      acc1 = __builtin_amdgcn_wmma_f32_16x16x32_bf16(false, a, false, frag_b_bf16(wrow1, k, half), (short)0, acc1, false, false);
      acc2 = __builtin_amdgcn_wmma_f32_16x16x32_bf16(false, a, false, frag_b_bf16(wrow2, k, half), (short)0, acc2, false, false);
      acc3 = __builtin_amdgcn_wmma_f32_16x16x32_bf16(false, a, false, frag_b_bf16(wrow3, k, half), (short)0, acc3, false, false);
    }
    if (c + 1 < 8) {
      __syncthreads();                               // all waves done reading buf c&1
      if (c + 2 < 8) {
        stage_chunk(xt, c & 1, Xb, m0, c + 2, tid);  // refill freed buffer
        wait_async1();                               // chunk c+1 arrived; c+2 in flight
      } else {
        wait_async0();
      }
      __syncthreads();
    }
  }
  v8f acc[4] = {acc0, acc1, acc2, acc3};

  const int h      = o_in >> 6;
  const int b_idx  = m0 >> 10;
  const int t_base = (m0 & (T_SEQ - 1)) + half * 8;

  if (proj < 2) {
    // RoPE: even/odd dim pairs live in adjacent lanes (dim parity == lane parity)
    __bf16* Obuf = (proj == 0) ? Qb : Kb;
    const float sgn = ((ln & 1) == 0) ? -1.f : 1.f;
#pragma unroll
    for (int nt = 0; nt < 4; nt++) {
      const int   dim      = nt * 16 + ln;
      const int   pair     = dim >> 1;
      const float inv_freq = exp2f(-(float)pair * (13.287712379549449f / 32.f));
#pragma unroll
      for (int r = 0; r < 8; r++) {
        float v       = acc[nt][r];
        float partner = __shfl_xor(v, 1, 32);
        int   t       = t_base + r;
        float th = (float)t * inv_freq, s, cs;
        __sincosf(th, &s, &cs);
        float out = v * cs + sgn * s * partner;
        Obuf[(((size_t)b_idx * NH + h) * T_SEQ + t) * DH + dim] = (__bf16)out;
      }
    }
  } else {
    // V transposed: lane owns a fixed dim, rows t_base..t_base+7 contiguous
#pragma unroll
    for (int nt = 0; nt < 4; nt++) {
      __align__(16) __bf16 tmp[8];
#pragma unroll
      for (int r = 0; r < 8; r++) tmp[r] = (__bf16)acc[nt][r];
      __bf16* dst = Vt + (((size_t)b_idx * NH + h) * DH + nt * 16 + ln) * T_SEQ + t_base;
      *(uint4*)dst = *(const uint4*)tmp;
    }
  }
}

// ---- flash helpers ------------------------------------------------------------
// Online-softmax update for one 16-query tile given its two 16-key score tiles.
// Lane owns query qg; its keys are K0+8*half+r (tile0) and K0+16+8*half+r (tile1);
// the xor-16 partner lane holds the complementary keys of the same query.
__device__ __forceinline__ v16bf softmax_update(const v8f& s0, const v8f& s1,
                                                int K0, int qg, int half,
                                                float& m_run, float& l_run,
                                                float& resc) {
  float sv[16], m_loc = -INFINITY;
#pragma unroll
  for (int r = 0; r < 8; r++) {
    int   k0i = K0 + 8 * half + r;
    int   k1i = K0 + 16 + 8 * half + r;
    float x0  = (k0i <= qg) ? s0[r] : -INFINITY;
    float x1  = (k1i <= qg) ? s1[r] : -INFINITY;
    sv[r] = x0; sv[8 + r] = x1;
    m_loc = fmaxf(m_loc, fmaxf(x0, x1));
  }
  float m_blk = fmaxf(m_loc, __shfl_xor(m_loc, 16, 32));
  float m_new = fmaxf(m_run, m_blk);
  resc = __expf(m_run - m_new);            // first block: exp(-inf)=0
  float lsum = 0.f;
  v16bf pa;                                 // already in A-fragment element order
#pragma unroll
  for (int i = 0; i < 16; i++) {
    float p = __expf(sv[i] - m_new);
    lsum += p;
    pa[i] = (__bf16)p;
  }
  lsum += __shfl_xor(lsum, 16, 32);
  l_run = l_run * resc + lsum;
  m_run = m_new;
  return pa;
}

// ---------------- Kernel 2: causal flash attention, one wave = 32 queries ------
__global__ __launch_bounds__(256)
void flash_attn_kernel(const __bf16* __restrict__ Qb,
                       const __bf16* __restrict__ Kb,
                       const __bf16* __restrict__ Vt,
                       __bf16* __restrict__ Ob) {  // [B,T,H*DH]
  const int lane = threadIdx.x & 31;
  const int wave = threadIdx.x >> 5;
  const int half = lane >> 4;
  const int ln   = lane & 15;

  const int bh = blockIdx.x;               // b*16+h
  const int qt = blockIdx.y * 8 + wave;    // 32-query tile index, 0..31
  const int qs = qt * 32;
  const int b  = bh >> 4, h = bh & 15;

  const __bf16* Qh = Qb + (size_t)bh * T_SEQ * DH;
  const __bf16* Kh = Kb + (size_t)bh * T_SEQ * DH;
  const __bf16* Vh = Vt + (size_t)bh * DH * T_SEQ;

  // Q^T as B-operand, fixed for the whole scan; two 16-query tiles A and B
  const __bf16* qrowA = Qh + (size_t)(qs + ln) * DH;
  const __bf16* qrowB = qrowA + (size_t)16 * DH;
  const v16bf qfA0 = frag_b_bf16(qrowA, 0, half), qfA1 = frag_b_bf16(qrowA, 32, half);
  const v16bf qfB0 = frag_b_bf16(qrowB, 0, half), qfB1 = frag_b_bf16(qrowB, 32, half);

  const int qgA = qs + ln, qgB = qs + 16 + ln;

  float mA = -INFINITY, lA = 0.f, mB = -INFINITY, lB = 0.f;
  v8f oA0 = {}, oA1 = {}, oA2 = {}, oA3 = {};
  v8f oB0 = {}, oB1 = {}, oB2 = {}, oB3 = {};

  for (int K0 = 0; K0 < qs + 32; K0 += 32) {
    // Shared K fragments for both query tiles
    const __bf16* krow0 = Kh + (size_t)(K0 + ln) * DH;
    const __bf16* krow1 = Kh + (size_t)(K0 + 16 + ln) * DH;
    const v16bf ka00 = frag_a_bf16(krow0, 0, half), ka01 = frag_a_bf16(krow0, 32, half);
    const v16bf ka10 = frag_a_bf16(krow1, 0, half), ka11 = frag_a_bf16(krow1, 32, half);

    v8f sA0 = {}, sA1 = {}, sB0 = {}, sB1 = {};
    sA0 = __builtin_amdgcn_wmma_f32_16x16x32_bf16(false, ka00, false, qfA0, (short)0, sA0, false, false);
    sA0 = __builtin_amdgcn_wmma_f32_16x16x32_bf16(false, ka01, false, qfA1, (short)0, sA0, false, false);
    sA1 = __builtin_amdgcn_wmma_f32_16x16x32_bf16(false, ka10, false, qfA0, (short)0, sA1, false, false);
    sA1 = __builtin_amdgcn_wmma_f32_16x16x32_bf16(false, ka11, false, qfA1, (short)0, sA1, false, false);
    sB0 = __builtin_amdgcn_wmma_f32_16x16x32_bf16(false, ka00, false, qfB0, (short)0, sB0, false, false);
    sB0 = __builtin_amdgcn_wmma_f32_16x16x32_bf16(false, ka01, false, qfB1, (short)0, sB0, false, false);
    sB1 = __builtin_amdgcn_wmma_f32_16x16x32_bf16(false, ka10, false, qfB0, (short)0, sB1, false, false);
    sB1 = __builtin_amdgcn_wmma_f32_16x16x32_bf16(false, ka11, false, qfB1, (short)0, sB1, false, false);

    // Shared V fragments (V^T rows: lane owns dim, keys contiguous)
    const v16bf vf0 = frag_b_bf16(Vh + (size_t)( 0 + ln) * T_SEQ, K0, half);
    const v16bf vf1 = frag_b_bf16(Vh + (size_t)(16 + ln) * T_SEQ, K0, half);
    const v16bf vf2 = frag_b_bf16(Vh + (size_t)(32 + ln) * T_SEQ, K0, half);
    const v16bf vf3 = frag_b_bf16(Vh + (size_t)(48 + ln) * T_SEQ, K0, half);

    // ---- tile A ----
    float rescA;
    v16bf paA = softmax_update(sA0, sA1, K0, qgA, half, mA, lA, rescA);
#pragma unroll
    for (int r = 0; r < 8; r++) {
      float sr = __shfl(rescA, r + 8 * half, 32);
      oA0[r] *= sr; oA1[r] *= sr; oA2[r] *= sr; oA3[r] *= sr;
    }
    oA0 = __builtin_amdgcn_wmma_f32_16x16x32_bf16(false, paA, false, vf0, (short)0, oA0, false, false);
    oA1 = __builtin_amdgcn_wmma_f32_16x16x32_bf16(false, paA, false, vf1, (short)0, oA1, false, false);
    oA2 = __builtin_amdgcn_wmma_f32_16x16x32_bf16(false, paA, false, vf2, (short)0, oA2, false, false);
    oA3 = __builtin_amdgcn_wmma_f32_16x16x32_bf16(false, paA, false, vf3, (short)0, oA3, false, false);

    // ---- tile B ----
    float rescB;
    v16bf paB = softmax_update(sB0, sB1, K0, qgB, half, mB, lB, rescB);
#pragma unroll
    for (int r = 0; r < 8; r++) {
      float sr = __shfl(rescB, r + 8 * half, 32);
      oB0[r] *= sr; oB1[r] *= sr; oB2[r] *= sr; oB3[r] *= sr;
    }
    oB0 = __builtin_amdgcn_wmma_f32_16x16x32_bf16(false, paB, false, vf0, (short)0, oB0, false, false);
    oB1 = __builtin_amdgcn_wmma_f32_16x16x32_bf16(false, paB, false, vf1, (short)0, oB1, false, false);
    oB2 = __builtin_amdgcn_wmma_f32_16x16x32_bf16(false, paB, false, vf2, (short)0, oB2, false, false);
    oB3 = __builtin_amdgcn_wmma_f32_16x16x32_bf16(false, paB, false, vf3, (short)0, oB3, false, false);
  }

  const float liA = 1.f / lA, liB = 1.f / lB;
#pragma unroll
  for (int r = 0; r < 8; r++) {
    float  sA  = __shfl(liA, r + 8 * half, 32);
    float  sB  = __shfl(liB, r + 8 * half, 32);
    int    tA  = qs + r + 8 * half;
    int    tB  = tA + 16;
    size_t bA  = ((size_t)b * T_SEQ + tA) * D_MODEL + h * DH;
    size_t bB  = ((size_t)b * T_SEQ + tB) * D_MODEL + h * DH;
    Ob[bA +  0 + ln] = (__bf16)(oA0[r] * sA);
    Ob[bA + 16 + ln] = (__bf16)(oA1[r] * sA);
    Ob[bA + 32 + ln] = (__bf16)(oA2[r] * sA);
    Ob[bA + 48 + ln] = (__bf16)(oA3[r] * sA);
    Ob[bB +  0 + ln] = (__bf16)(oB0[r] * sB);
    Ob[bB + 16 + ln] = (__bf16)(oB1[r] * sB);
    Ob[bB + 32 + ln] = (__bf16)(oB2[r] * sB);
    Ob[bB + 48 + ln] = (__bf16)(oB3[r] * sB);
  }
}

// ---------------- Kernel 3: output projection (LDS-staged A, 16x64/wave) -------
__global__ __launch_bounds__(256)
void out_proj_kernel(const __bf16* __restrict__ Ob,
                     const __bf16* __restrict__ Wob,
                     float* __restrict__ out) {
  __shared__ __align__(16) __bf16 xt[2][16][XSTRIDE];

  const int tid  = threadIdx.x;
  const int lane = tid & 31;
  const int wave = tid >> 5;
  const int half = lane >> 4;
  const int ln   = lane & 15;

  const int m0 = blockIdx.x * 16;
  const int og = (blockIdx.y * 8 + wave) * 64;

  const __bf16* wrow0 = Wob + (size_t)(og +  0 + ln) * D_MODEL;
  const __bf16* wrow1 = Wob + (size_t)(og + 16 + ln) * D_MODEL;
  const __bf16* wrow2 = Wob + (size_t)(og + 32 + ln) * D_MODEL;
  const __bf16* wrow3 = Wob + (size_t)(og + 48 + ln) * D_MODEL;

  stage_chunk(xt, 0, Ob, m0, 0, tid);
  wait_async0();
  __syncthreads();
  stage_chunk(xt, 1, Ob, m0, 1, tid);

  v8f acc0 = {}, acc1 = {}, acc2 = {}, acc3 = {};
  for (int c = 0; c < 8; c++) {
    const __bf16* arow = &xt[c & 1][ln][0];
#pragma unroll
    for (int kk = 0; kk < KC; kk += 32) {
      const int k = c * KC + kk;
      v16bf a = frag_a_bf16(arow, kk, half);
      acc0 = __builtin_amdgcn_wmma_f32_16x16x32_bf16(false, a, false, frag_b_bf16(wrow0, k, half), (short)0, acc0, false, false);
      acc1 = __builtin_amdgcn_wmma_f32_16x16x32_bf16(false, a, false, frag_b_bf16(wrow1, k, half), (short)0, acc1, false, false);
      acc2 = __builtin_amdgcn_wmma_f32_16x16x32_bf16(false, a, false, frag_b_bf16(wrow2, k, half), (short)0, acc2, false, false);
      acc3 = __builtin_amdgcn_wmma_f32_16x16x32_bf16(false, a, false, frag_b_bf16(wrow3, k, half), (short)0, acc3, false, false);
    }
    if (c + 1 < 8) {
      __syncthreads();
      if (c + 2 < 8) {
        stage_chunk(xt, c & 1, Ob, m0, c + 2, tid);
        wait_async1();
      } else {
        wait_async0();
      }
      __syncthreads();
    }
  }
  v8f acc[4] = {acc0, acc1, acc2, acc3};
#pragma unroll
  for (int nt = 0; nt < 4; nt++)
#pragma unroll
    for (int r = 0; r < 8; r++)
      out[(size_t)(m0 + r + 8 * half) * D_MODEL + og + nt * 16 + ln] = acc[nt][r];
}

extern "C" void kernel_launch(void* const* d_in, const int* in_sizes, int n_in,
                              void* d_out, int out_size, void* d_ws, size_t ws_size,
                              hipStream_t stream) {
  const float* x  = (const float*)d_in[0];
  const float* Wq = (const float*)d_in[1];
  const float* Wk = (const float*)d_in[2];
  const float* Wv = (const float*)d_in[3];
  const float* Wo = (const float*)d_in[4];

  const size_t nx = (size_t)NBATCH * T_SEQ * D_MODEL;          // 4 Mi
  const size_t nw = (size_t)D_MODEL * D_MODEL;                 // 1 Mi
  const size_t na = (size_t)NBATCH * NH * T_SEQ * DH;          // 4 Mi

  __bf16* Xb  = (__bf16*)d_ws;
  __bf16* Wqb = Xb  + nx;
  __bf16* Wkb = Wqb + nw;
  __bf16* Wvb = Wkb + nw;
  __bf16* Wob = Wvb + nw;
  __bf16* Qb  = Wob + nw;
  __bf16* Kb  = Qb  + na;
  __bf16* Vt  = Kb  + na;
  __bf16* Ob  = Vt  + na;   // total 24 Mi bf16 = 48 MiB

  dim3 blk(256);
  cvt_bf16_kernel<<<dim3((int)(nx / 4 / 256)), blk, 0, stream>>>(x,  Xb,  (int)(nx / 4));
  cvt_bf16_kernel<<<dim3((int)(nw / 4 / 256)), blk, 0, stream>>>(Wq, Wqb, (int)(nw / 4));
  cvt_bf16_kernel<<<dim3((int)(nw / 4 / 256)), blk, 0, stream>>>(Wk, Wkb, (int)(nw / 4));
  cvt_bf16_kernel<<<dim3((int)(nw / 4 / 256)), blk, 0, stream>>>(Wv, Wvb, (int)(nw / 4));
  cvt_bf16_kernel<<<dim3((int)(nw / 4 / 256)), blk, 0, stream>>>(Wo, Wob, (int)(nw / 4));

  // 256 row tiles x 48 col tiles of 64 (Q|K|V), 8 waves/block
  qkv_rope_kernel<<<dim3(256, 6), blk, 0, stream>>>(Xb, Wqb, Wkb, Wvb, Qb, Kb, Vt);
  // 64 (b,h) x 32 query tiles of 32, 8 waves/block
  flash_attn_kernel<<<dim3(64, 4), blk, 0, stream>>>(Qb, Kb, Vt, Ob);
  // 256 row tiles x 16 col tiles of 64
  out_proj_kernel<<<dim3(256, 2), blk, 0, stream>>>(Ob, Wob, (float*)d_out);
}
